// InformationExtractor_26560077758598
// MI455X (gfx1250) — matrix operand order repaired and checked by reading, not compiled, not run on hardware
//
#include <hip/hip_runtime.h>
#include <stdint.h>

typedef __attribute__((ext_vector_type(16))) __bf16 v16bf;
typedef __attribute__((ext_vector_type(8)))  __bf16 v8bf;
typedef __attribute__((ext_vector_type(8)))  float  v8f;
typedef __attribute__((ext_vector_type(4)))  int    v4i;

#define KDIM 288    // K = 9 taps * 32 channels, ordered k = tap*32 + c
#define KSTR 296    // padded LDS stride (halfs)
#define MTILE 128   // M rows per block

#if defined(__has_builtin)
#if __has_builtin(__builtin_amdgcn_global_load_async_to_lds_b128) && \
    __has_builtin(__builtin_amdgcn_s_wait_asynccnt)
#define USE_ASYNC 1
#endif
#endif

#ifdef USE_ASYNC
typedef __attribute__((address_space(1))) v4i GlbV4;   // global (AS1)
typedef __attribute__((address_space(3))) v4i LdsV4;   // LDS (AS3)
#endif

// 16-byte global -> LDS copy; async on gfx1250 toolchains that expose it.
__device__ __forceinline__ void copy16_g2l(__bf16* dst, const __bf16* src)
{
#ifdef USE_ASYNC
    GlbV4* g = (GlbV4*)(uintptr_t)src;
    LdsV4* l = (LdsV4*)(uintptr_t)dst;
    __builtin_amdgcn_global_load_async_to_lds_b128(g, l, 0, 0);
#else
    *(v4i*)dst = *(const v4i*)src;
#endif
}

// ---------------------------------------------------------------------------
// Kernel 1: conv1 3x3 SAME (1->32) + ReLU + 2x2 maxpool.
// Output bf16 NHWC [B,14,14,32]. Thread order: channel fastest, so each
// wave's 32 lanes share one (b, pixel) -> readfirstlane makes all image-tap
// addresses wave-uniform (scalar s_load path), only weights differ per lane.
// ---------------------------------------------------------------------------
__global__ __launch_bounds__(256) void k_conv1_pool(
    const float* __restrict__ x, const float* __restrict__ w,
    const float* __restrict__ bias, __bf16* __restrict__ f1, int B)
{
    int idx = blockIdx.x * blockDim.x + threadIdx.x;
    if (idx >= B * 196 * 32) return;
    int c  = idx & 31;
    int g  = __builtin_amdgcn_readfirstlane(idx >> 5);  // uniform: b*196 + pp
    int pp = g % 196;
    int b  = g / 196;
    int y  = pp / 14, xo = pp % 14;

    float wv[9];
#pragma unroll
    for (int i = 0; i < 9; ++i) wv[i] = w[c * 9 + i];
    float bv = bias[c];
    const float* xb = x + (size_t)b * 784;

    float mx = 0.0f;                // relu(max(..)) == max(0, raw convs)
#pragma unroll
    for (int dy = 0; dy < 2; ++dy) {
#pragma unroll
        for (int dx = 0; dx < 2; ++dx) {
            int oy = 2 * y + dy, ox = 2 * xo + dx;
            float acc = bv;
#pragma unroll
            for (int r = 0; r < 3; ++r) {
                int iy = oy + r - 1;
                if (iy < 0 || iy >= 28) continue;
#pragma unroll
                for (int s = 0; s < 3; ++s) {
                    int ix = ox + s - 1;
                    if (ix < 0 || ix >= 28) continue;
                    acc = fmaf(xb[iy * 28 + ix], wv[r * 3 + s], acc);
                }
            }
            mx = fmaxf(mx, acc);
        }
    }
    f1[idx] = (__bf16)mx;
}

// ---------------------------------------------------------------------------
// Kernel 2: pack conv2_w [64][32][3][3] f32 into bf16 per-lane WMMA B
// fragments: W2f[((nt*9 + kk)*32 + lane)*16 + j]
//   n = nt*16 + (lane&15), k = kk*32 + (lane>>4)*16 + j, k = tap*32 + c.
// ---------------------------------------------------------------------------
__global__ __launch_bounds__(256) void k_pack_w2(
    const float* __restrict__ w, __bf16* __restrict__ W2f)
{
    int i = blockIdx.x * blockDim.x + threadIdx.x;
    if (i >= 64 * KDIM) return;
    int j    = i & 15;
    int lane = (i >> 4) & 31;
    int kkq  = i >> 9;
    int kk   = kkq % 9;
    int nt   = kkq / 9;
    int n    = nt * 16 + (lane & 15);
    int k    = kk * 32 + (lane >> 4) * 16 + j;
    int tap  = k >> 5, c = k & 31;
    W2f[i] = (__bf16)w[(n * 32 + c) * 9 + tap];
}

__device__ __forceinline__ v16bf ldfrag(const __bf16* p, int stride)
{
    v8bf lo = *(const v8bf*)(p);
    v8bf hi = *(const v8bf*)(p + stride);
    return __builtin_shufflevector(lo, hi,
        0,1,2,3,4,5,6,7,8,9,10,11,12,13,14,15);
}

// ---------------------------------------------------------------------------
// Kernel 3: conv2 as implicit GEMM [M=B*196, K=288] x [K=288, N=64].
// Block: 256 threads = 8 wave32, tile 128(M) x 64(N).
// Each wave: 2(M) x 2(N) 16x16 sub-tiles -> every A fragment feeds 2 WMMAs
// (half the LDS traffic per WMMA), B fragments in registers from packed
// global weights. 36 WMMAs/wave. A tile staged via async global->LDS b128.
// ---------------------------------------------------------------------------
__global__ __launch_bounds__(256) void k_conv2_wmma(
    const __bf16* __restrict__ f1, const __bf16* __restrict__ W2f,
    const float* __restrict__ bias, __bf16* __restrict__ c2, int B)
{
    __shared__ __bf16 As[MTILE * KSTR];

    const int tid = threadIdx.x;
    const int M0  = blockIdx.x * MTILE;

    __builtin_prefetch(W2f, 0, 1);   // global_prefetch_b8

    // Stage im2col A tile: 128 rows * 9 taps * 4 chunks(16B) = 4608 chunks,
    // 18 per thread. In-bounds -> async global->LDS; padding -> zero store.
#pragma unroll
    for (int t = 0; t < 18; ++t) {
        int cid = tid + t * 256;
        int q   = cid & 3;             // 8-half chunk within the 32-chan row
        int tap = (cid >> 2) % 9;
        int row = cid / 36;
        int m   = M0 + row;
        int p   = m % 196, b = m / 196;
        int y   = p / 14,  xx = p % 14;
        int r   = tap / 3, s = tap % 3;
        int iy  = y + r - 1, ix = xx + s - 1;
        __bf16* dst = &As[row * KSTR + tap * 32 + q * 8];
        if (iy >= 0 && iy < 14 && ix >= 0 && ix < 14) {
            const __bf16* src =
                f1 + (((size_t)b * 196 + iy * 14 + ix) << 5) + q * 8;
            copy16_g2l(dst, src);
        } else {
            *(v4i*)dst = (v4i){0, 0, 0, 0};
        }
    }
#ifdef USE_ASYNC
    __builtin_amdgcn_s_wait_asynccnt(0);
#endif
    __syncthreads();

    const int lane = tid & 31;
    const int wv   = tid >> 5;
    const int mtp  = wv & 3;     // M pair 0..3  -> rows [mtp*32, mtp*32+32)
    const int ntp  = wv >> 2;    // N pair 0..1  -> cols [ntp*32, ntp*32+32)
    const int lh   = lane >> 4;
    const int l15  = lane & 15;

    v8f acc00 = {}, acc01 = {}, acc10 = {}, acc11 = {};
    const __bf16* wb0 = W2f + (size_t)(2 * ntp * 9) * 512;  // nt = 2*ntp
    const __bf16* wb1 = wb0 + (size_t)9 * 512;              // nt = 2*ntp+1
    const int ab0 = (mtp * 32 + l15) * KSTR + (lh ? 16 : 0);
    const int ab1 = ab0 + 16 * KSTR;

#pragma unroll
    for (int kk = 0; kk < 9; ++kk) {
        const int wo = kk * 512 + lane * 16;
        v16bf b0 = ldfrag(wb0 + wo, 8);
        v16bf b1 = ldfrag(wb1 + wo, 8);
        const int ka = kk * 32;
        v16bf a0 = ldfrag(&As[ab0 + ka], 8);
        v16bf a1 = ldfrag(&As[ab1 + ka], 8);

        acc00 = __builtin_amdgcn_wmma_f32_16x16x32_bf16(
            false, a0, false, b0, (short)0, acc00, false, false);
        acc01 = __builtin_amdgcn_wmma_f32_16x16x32_bf16(
            false, a0, false, b1, (short)0, acc01, false, false);
        acc10 = __builtin_amdgcn_wmma_f32_16x16x32_bf16(
            false, a1, false, b0, (short)0, acc10, false, false);
        acc11 = __builtin_amdgcn_wmma_f32_16x16x32_bf16(
            false, a1, false, b1, (short)0, acc11, false, false);
    }

    // Epilogue: lanes 0-15: m = v, lanes 16-31: m = v+8; n = lane%16.
    // One m -> (b,p) division feeds both N sub-tiles.
    const int n0 = ntp * 32 + l15;
    const int n1 = n0 + 16;
    const float bv0 = bias[n0];
    const float bv1 = bias[n1];
#pragma unroll
    for (int mti = 0; mti < 2; ++mti) {
        v8f aN0 = mti ? acc10 : acc00;
        v8f aN1 = mti ? acc11 : acc01;
        const int mb = M0 + mtp * 32 + mti * 16 + 8 * lh;
#pragma unroll
        for (int v = 0; v < 8; ++v) {
            int m = mb + v;
            int p = m % 196, b = m / 196;
            size_t base = ((size_t)b * 64) * 196 + p;
            c2[base + (size_t)n0 * 196] = (__bf16)fmaxf(aN0[v] + bv0, 0.0f);
            c2[base + (size_t)n1 * 196] = (__bf16)fmaxf(aN1[v] + bv1, 0.0f);
        }
    }
}

// ---------------------------------------------------------------------------
// Kernel 4: per-sample tail (pool + sigmoid attention + experts + final).
// 64 threads / sample; thread = expert/channel.
// relu(sum (feat*attn)*w + b) == relu(attn*dot(feat,w) + b)
// ---------------------------------------------------------------------------
__global__ __launch_bounds__(64) void k_tail(
    const __bf16* __restrict__ c2,
    const float* __restrict__ attn_w, const float* __restrict__ attn_bp,
    const float* __restrict__ ew, const float* __restrict__ eb,
    const float* __restrict__ fw, const float* __restrict__ fb,
    float* __restrict__ out)
{
    __shared__ float aw[49];
    __shared__ float eo[128];
    const int e = threadIdx.x;
    const int b = blockIdx.x;

    if (e < 49) aw[e] = attn_w[e];
    __syncthreads();

    const __bf16* ch = c2 + ((size_t)b * 64 + e) * 196;
    float feat[49];
    float dot = 0.0f;
#pragma unroll
    for (int f = 0; f < 49; ++f) {
        int y = f / 7, x = f % 7;
        const __bf16* r0 = ch + (2 * y) * 14 + 2 * x;
        const __bf16* r1 = r0 + 14;
        float v = fmaxf(fmaxf((float)r0[0], (float)r0[1]),
                        fmaxf((float)r1[0], (float)r1[1]));
        feat[f] = v;
        dot = fmaf(v, aw[f], dot);
    }
    float attn = 1.0f / (1.0f + __expf(-(dot + attn_bp[0])));

#pragma unroll
    for (int o = 0; o < 2; ++o) {
        float s = 0.0f;
#pragma unroll
        for (int f = 0; f < 49; ++f)
            s = fmaf(feat[f], ew[(e * 49 + f) * 2 + o], s);
        eo[e * 2 + o] = fmaxf(attn * s + eb[e * 2 + o], 0.0f);
    }
    __syncthreads();

    if (e < 2) {
        float s = fb[e];
#pragma unroll 8
        for (int i = 0; i < 128; ++i)
            s = fmaf(eo[i], fw[i * 2 + e], s);
        out[(size_t)b * 2 + e] = s;
    }
}

// ---------------------------------------------------------------------------
extern "C" void kernel_launch(void* const* d_in, const int* in_sizes, int n_in,
                              void* d_out, int out_size, void* d_ws, size_t ws_size,
                              hipStream_t stream)
{
    const float* x   = (const float*)d_in[0];
    const float* c1w = (const float*)d_in[1];
    const float* c1b = (const float*)d_in[2];
    const float* c2w = (const float*)d_in[3];
    const float* c2b = (const float*)d_in[4];
    const float* aw  = (const float*)d_in[5];
    const float* ab  = (const float*)d_in[6];
    const float* ew  = (const float*)d_in[7];
    const float* eb  = (const float*)d_in[8];
    const float* fw  = (const float*)d_in[9];
    const float* fb  = (const float*)d_in[10];
    float* out = (float*)d_out;

    const int B = in_sizes[0] / 784;   // x is [B,1,28,28]

    char* ws = (char*)d_ws;
    size_t f1_bytes = (size_t)B * 196 * 32 * 2;                  // bf16 NHWC
    size_t w2_bytes = ((size_t)64 * KDIM * 2 + 255) & ~(size_t)255;
    __bf16* f1  = (__bf16*)(ws);
    __bf16* W2f = (__bf16*)(ws + f1_bytes);
    __bf16* c2  = (__bf16*)(ws + f1_bytes + w2_bytes);           // bf16 NCHW

    const int n1 = B * 196 * 32;
    k_conv1_pool<<<(n1 + 255) / 256, 256, 0, stream>>>(x, c1w, c1b, f1, B);
    k_pack_w2<<<(64 * KDIM + 255) / 256, 256, 0, stream>>>(c2w, W2f);

    const int M = B * 196;                    // divisible by MTILE=128
    k_conv2_wmma<<<M / MTILE, 256, 0, stream>>>(f1, W2f, c2b, c2, B);

    k_tail<<<B, 64, 0, stream>>>(c2, aw, ab, ew, eb, fw, fb, out);
}
